// RWKV_GPT_71906342470238
// MI455X (gfx1250) — compile-verified, matrix operand order checked by reading, not run on hardware
//
#include <hip/hip_runtime.h>
#include <hip/hip_bf16.h>
#include <stdint.h>

// ---------------- model dims ----------------
#define TT 1024
#define CC 768
#define HH 12
#define HD 64
#define FF 3072
#define VV 50257
#define NL 4
#define DDEC 64
#define DAAA 64
#define DMV  32
#define DGATE 128

typedef __attribute__((ext_vector_type(16))) __bf16 bf16x16;
typedef __attribute__((ext_vector_type(8)))  float  f32x8;
typedef __attribute__((ext_vector_type(4)))  int    i32x4;
typedef __attribute__((address_space(1))) i32x4 i32x4_g;   // global
typedef __attribute__((address_space(3))) i32x4 i32x4_l;   // LDS

#if defined(__has_builtin)
#if __has_builtin(__builtin_amdgcn_global_load_async_to_lds_b128)
#define HAVE_ASYNC_LDS 1
#endif
#endif
#ifndef HAVE_ASYNC_LDS
#define HAVE_ASYNC_LDS 0
#endif

// ================= conversion kernels =================
__global__ __launch_bounds__(256) void k_cvt(const float* __restrict__ in,
                                             __bf16* __restrict__ out, int n) {
    int i = blockIdx.x * 256 + threadIdx.x;
    if (i < n) out[i] = (__bf16)in[i];
}

// W[K,N] f32 -> Wt[N,K] bf16 (transpose so B-columns are K-contiguous)
__global__ __launch_bounds__(256) void k_cvt_t(const float* __restrict__ W,
                                               __bf16* __restrict__ Wt, int K, int N) {
    int i = blockIdx.x * 256 + threadIdx.x;
    if (i >= K * N) return;
    int k = i / N, n = i - k * N;
    Wt[(size_t)n * K + k] = (__bf16)W[i];
}

// ================= WMMA GEMM =================
// C[M,N] = A[M,K] * B[K,N]; A bf16 row-major [M,K]; Bt = B^T bf16 [N,K].
// M % 128 == 0, K % 32 == 0, N arbitrary (guarded).
// 256 threads = 8 waves; macro tile 128x128; wave computes 2x4 16x16 tiles;
// double-buffered LDS fed by async global->LDS when available.
#define BM 128
#define BN 128
#define BK 32
#define LSTR 40   // bf16 elems per LDS row: 64B data + 16B pad (16B aligned chunks)

__global__ __launch_bounds__(256) void k_gemm(const __bf16* __restrict__ A,
                                              const __bf16* __restrict__ Bt,
                                              float* __restrict__ C,
                                              int M, int N, int K) {
    __shared__ __align__(16) __bf16 As[2][BM * LSTR];
    __shared__ __align__(16) __bf16 Bs[2][BN * LSTR];
    const int tid  = threadIdx.x;
    const int bm   = blockIdx.x * BM;
    const int bn   = blockIdx.y * BN;
    const int wave = tid >> 5, lane = tid & 31;
    const int wr   = wave & 3;          // row group: 32 rows
    const int wc   = wave >> 2;         // col group: 64 cols
    const int hl   = lane & 15;
    const int klo  = (lane >> 4) * 8;   // lanes 16-31 take K halves +8

    // staging assignment: each thread owns two 16B chunks of one row
    const int crow   = tid >> 1;            // 0..127
    const int cchunk = (tid & 1) * 2;       // chunk pair {0,1} or {2,3}
    const int arow   = bm + crow;
    const int brow   = bn + crow;
    const bool bvalid = (brow < N);

    // pre-zero ragged B rows once (both buffers); async loads skip them
    if (bn + BN > N && !bvalid) {
        float4 z = {0.f, 0.f, 0.f, 0.f};
        *(float4*)&Bs[0][crow * LSTR + cchunk * 8]       = z;
        *(float4*)&Bs[0][crow * LSTR + (cchunk + 1) * 8] = z;
        *(float4*)&Bs[1][crow * LSTR + cchunk * 8]       = z;
        *(float4*)&Bs[1][crow * LSTR + (cchunk + 1) * 8] = z;
    }

#if HAVE_ASYNC_LDS
    auto issue = [&](int buf, int k0) {
        {
            const __bf16* g = A + (size_t)arow * K + k0 + cchunk * 8;
            __bf16* l = &As[buf][crow * LSTR + cchunk * 8];
            __builtin_amdgcn_global_load_async_to_lds_b128(
                (i32x4_g*)g, (i32x4_l*)l, 0, 0);
            __builtin_amdgcn_global_load_async_to_lds_b128(
                (i32x4_g*)(g + 8), (i32x4_l*)(l + 8), 0, 0);
        }
        if (bvalid) {
            const __bf16* g = Bt + (size_t)brow * K + k0 + cchunk * 8;
            __bf16* l = &Bs[buf][crow * LSTR + cchunk * 8];
            __builtin_amdgcn_global_load_async_to_lds_b128(
                (i32x4_g*)g, (i32x4_l*)l, 0, 0);
            __builtin_amdgcn_global_load_async_to_lds_b128(
                (i32x4_g*)(g + 8), (i32x4_l*)(l + 8), 0, 0);
        }
    };
#else
    auto stage = [&](int buf, int k0) {
        {
            const __bf16* g = A + (size_t)arow * K + k0 + cchunk * 8;
            __bf16* l = &As[buf][crow * LSTR + cchunk * 8];
            *(float4*)l       = *(const float4*)g;
            *(float4*)(l + 8) = *(const float4*)(g + 8);
        }
        {
            __bf16* l = &Bs[buf][crow * LSTR + cchunk * 8];
            if (bvalid) {
                const __bf16* g = Bt + (size_t)brow * K + k0 + cchunk * 8;
                *(float4*)l       = *(const float4*)g;
                *(float4*)(l + 8) = *(const float4*)(g + 8);
            } else {
                float4 z = {0.f, 0.f, 0.f, 0.f};
                *(float4*)l = z; *(float4*)(l + 8) = z;
            }
        }
    };
#endif

    f32x8 acc[2][4] = {};
    const int nk = K / BK;

#if HAVE_ASYNC_LDS
    issue(0, 0);
    asm volatile("s_wait_asynccnt 0" ::: "memory");
#else
    stage(0, 0);
#endif
    __syncthreads();

    for (int s = 0; s < nk; ++s) {
        const int cur = s & 1;
#if HAVE_ASYNC_LDS
        if (s + 1 < nk) issue(cur ^ 1, (s + 1) * BK);
#endif
        bf16x16 af[2], bfr[4];
        #pragma unroll
        for (int i = 0; i < 2; ++i) {
            const __bf16* p = &As[cur][(wr * 32 + i * 16 + hl) * LSTR];
            #pragma unroll
            for (int e = 0; e < 8; ++e) {
                af[i][e]     = p[klo + e];
                af[i][e + 8] = p[16 + klo + e];
            }
        }
        #pragma unroll
        for (int j = 0; j < 4; ++j) {
            const __bf16* p = &Bs[cur][(wc * 64 + j * 16 + hl) * LSTR];
            #pragma unroll
            for (int e = 0; e < 8; ++e) {
                bfr[j][e]     = p[klo + e];
                bfr[j][e + 8] = p[16 + klo + e];
            }
        }
        #pragma unroll
        for (int i = 0; i < 2; ++i)
            #pragma unroll
            for (int j = 0; j < 4; ++j)
                acc[i][j] = __builtin_amdgcn_wmma_f32_16x16x32_bf16(
                    false, af[i], false, bfr[j], (short)0, acc[i][j], false, false);
#if HAVE_ASYNC_LDS
        if (s + 1 < nk) asm volatile("s_wait_asynccnt 0" ::: "memory");
        __syncthreads();
#else
        __syncthreads();
        if (s + 1 < nk) { stage(cur ^ 1, (s + 1) * BK); __syncthreads(); }
#endif
    }

    // D layout: VGPR v -> row v (+8 for lanes>=16), col = lane&15
    #pragma unroll
    for (int i = 0; i < 2; ++i) {
        const int rbase = bm + wr * 32 + i * 16 + ((lane >> 4) * 8);
        #pragma unroll
        for (int j = 0; j < 4; ++j) {
            const int c = bn + wc * 64 + j * 16 + hl;
            if (c < N) {
                #pragma unroll
                for (int v = 0; v < 8; ++v)
                    C[(size_t)(rbase + v) * N + c] = acc[i][j][v];
            }
        }
    }
}

// ================= elementwise / small kernels =================
__global__ __launch_bounds__(256) void k_embed(const float* __restrict__ emb,
                                               const int* __restrict__ idx,
                                               float* __restrict__ x) {
    int i = blockIdx.x * 256 + threadIdx.x;
    if (i >= TT * CC) return;
    int t = i / CC, c = i - t * CC;
    x[i] = emb[(size_t)idx[t] * CC + c];
}

// LayerNorm over C=768; one block (256 threads) per token; safe in-place.
__global__ __launch_bounds__(256) void k_ln(const float* __restrict__ x,
                                            const float* __restrict__ g,
                                            const float* __restrict__ b,
                                            float* __restrict__ out) {
    __shared__ float red[256];
    int t = blockIdx.x, tid = threadIdx.x;
    const float* row = x + (size_t)t * CC;
    float v0 = row[tid], v1 = row[tid + 256], v2 = row[tid + 512];
    red[tid] = v0 + v1 + v2; __syncthreads();
    for (int s = 128; s > 0; s >>= 1) { if (tid < s) red[tid] += red[tid + s]; __syncthreads(); }
    float mu = red[0] * (1.0f / CC); __syncthreads();
    float d0 = v0 - mu, d1 = v1 - mu, d2 = v2 - mu;
    red[tid] = d0 * d0 + d1 * d1 + d2 * d2; __syncthreads();
    for (int s = 128; s > 0; s >>= 1) { if (tid < s) red[tid] += red[tid + s]; __syncthreads(); }
    float rstd = rsqrtf(red[0] * (1.0f / CC) + 1e-5f);
    float* o = out + (size_t)t * CC;
    o[tid]       = d0 * rstd * g[tid]       + b[tid];
    o[tid + 256] = d1 * rstd * g[tid + 256] + b[tid + 256];
    o[tid + 512] = d2 * rstd * g[tid + 512] + b[tid + 512];
}

// token-shift mix: out = x + (x[t-1]-x)*coef   (B=1)
__global__ __launch_bounds__(256) void k_mix(const float* __restrict__ x,
                                             const float* __restrict__ coef,
                                             float* __restrict__ out) {
    int i = blockIdx.x * 256 + threadIdx.x;
    if (i >= TT * CC) return;
    int c = i % CC;
    float xv = x[i];
    float xp = (i >= CC) ? x[i - CC] : 0.0f;
    out[i] = xv + (xp - xv) * coef[c];
}

__global__ __launch_bounds__(256) void k_tanh(float* x, int n) {
    int i = blockIdx.x * 256 + threadIdx.x;
    if (i < n) x[i] = tanhf(x[i]);
}
__global__ __launch_bounds__(256) void k_sigmoid(float* x, int n) {
    int i = blockIdx.x * 256 + threadIdx.x;
    if (i < n) x[i] = 1.0f / (1.0f + expf(-x[i]));
}
__global__ __launch_bounds__(256) void k_relusq(float* x, int n) {
    int i = blockIdx.x * 256 + threadIdx.x;
    if (i < n) { float r = fmaxf(x[i], 0.0f); x[i] = r * r; }
}
__global__ __launch_bounds__(256) void k_add(float* __restrict__ x,
                                             const float* __restrict__ d, int n) {
    int i = blockIdx.x * 256 + threadIdx.x;
    if (i < n) x[i] += d[i];
}
// w = -softplus(-(w0 + u)) - 0.5   (in-place on u=w buffer)
__global__ __launch_bounds__(256) void k_decay(float* __restrict__ w,
                                               const float* __restrict__ w0) {
    int i = blockIdx.x * 256 + threadIdx.x;
    if (i >= TT * CC) return;
    float u = w0[i % CC] + w[i];
    w[i] = -log1pf(expf(-u)) - 0.5f;
}
// a = sigmoid(a0 + u)   (in-place)
__global__ __launch_bounds__(256) void k_afin(float* __restrict__ a,
                                              const float* __restrict__ a0) {
    int i = blockIdx.x * 256 + threadIdx.x;
    if (i >= TT * CC) return;
    a[i] = 1.0f / (1.0f + expf(-(a0[i % CC] + a[i])));
}
// v = v + (vf - v) * sigmoid(v0 + u)
__global__ __launch_bounds__(256) void k_vmix(float* __restrict__ v,
                                              const float* __restrict__ vf,
                                              const float* __restrict__ v0,
                                              const float* __restrict__ u) {
    int i = blockIdx.x * 256 + threadIdx.x;
    if (i >= TT * CC) return;
    float s = 1.0f / (1.0f + expf(-(v0[i % CC] + u[i])));
    float vv = v[i];
    v[i] = vv + (vf[i] - vv) * s;
}

// kk prep: per (t,h), 64 threads. kk=k*k_k; L2-normalize per head;
// nkk=-kkn; kka=kkn*a; kout=k*(1+(a-1)*k_a)
__global__ __launch_bounds__(64) void k_kkprep(const float* __restrict__ k,
                                               const float* __restrict__ a,
                                               const float* __restrict__ kkc,
                                               const float* __restrict__ kac,
                                               float* __restrict__ nkk,
                                               float* __restrict__ kka,
                                               float* __restrict__ kout) {
    __shared__ float red[HD];
    int t = blockIdx.x, h = blockIdx.y, i = threadIdx.x;
    int c = h * HD + i;
    size_t off = (size_t)t * CC + c;
    float kv = k[off], av = a[off];
    float kk = kv * kkc[c];
    red[i] = kk * kk; __syncthreads();
    for (int s = 32; s > 0; s >>= 1) { if (i < s) red[i] += red[i + s]; __syncthreads(); }
    float inv = 1.0f / fmaxf(sqrtf(red[0]), 1e-12f);
    float kkn = kk * inv;
    nkk[off]  = -kkn;
    kka[off]  = kkn * av;
    kout[off] = kv * (1.0f + (av - 1.0f) * kac[c]);
}

// WKV7 scan: one block per head, 64 threads; thread i owns state row S[i][*]
__global__ __launch_bounds__(64) void k_wkv(const float* __restrict__ r,
                                            const float* __restrict__ w,
                                            const float* __restrict__ k,
                                            const float* __restrict__ v,
                                            const float* __restrict__ a,
                                            const float* __restrict__ b,
                                            float* __restrict__ y) {
    __shared__ float lr[HD], lw[HD], lk[HD], lv[HD], la[HD], lb[HD];
    int h = blockIdx.x, i = threadIdx.x;
    int base = h * HD + i;
    float S[HD];
    #pragma unroll
    for (int j = 0; j < HD; ++j) S[j] = 0.0f;
    for (int t = 0; t < TT; ++t) {
        size_t off = (size_t)t * CC + base;
        lr[i] = r[off];
        lw[i] = expf(-expf(w[off]));   // decay in (0,1)
        lk[i] = k[off];
        lv[i] = v[off];
        la[i] = a[off];
        lb[i] = b[off];
        __syncthreads();
        float sa = 0.0f;
        #pragma unroll
        for (int j = 0; j < HD; ++j) sa += S[j] * la[j];
        float vi = lv[i];
        float yi = 0.0f;
        #pragma unroll
        for (int j = 0; j < HD; ++j) {
            float s = S[j] * lw[j] + sa * lb[j] + vi * lk[j];
            S[j] = s;
            yi += s * lr[j];
        }
        y[off] = yi;
        __syncthreads();
    }
}

// GroupNorm(head) + bonus + gate; per (t,h), 64 threads; in-place on y
__global__ __launch_bounds__(64) void k_gnbg(float* __restrict__ y,
                                             const float* __restrict__ gw,
                                             const float* __restrict__ gb,
                                             const float* __restrict__ r,
                                             const float* __restrict__ k,
                                             const float* __restrict__ v,
                                             const float* __restrict__ rk,
                                             const float* __restrict__ g) {
    __shared__ float red[HD];
    int t = blockIdx.x, h = blockIdx.y, i = threadIdx.x;
    int c = h * HD + i;
    size_t off = (size_t)t * CC + c;
    float yv = y[off];
    red[i] = yv; __syncthreads();
    for (int s = 32; s > 0; s >>= 1) { if (i < s) red[i] += red[i + s]; __syncthreads(); }
    float mu = red[0] * (1.0f / HD); __syncthreads();
    float d = yv - mu;
    red[i] = d * d; __syncthreads();
    for (int s = 32; s > 0; s >>= 1) { if (i < s) red[i] += red[i + s]; __syncthreads(); }
    float gn = d * rsqrtf(red[0] * (1.0f / HD) + 64e-5f) * gw[c] + gb[c];
    __syncthreads();
    red[i] = r[off] * k[off] * rk[h * HD + i]; __syncthreads();
    for (int s = 32; s > 0; s >>= 1) { if (i < s) red[i] += red[i + s]; __syncthreads(); }
    float bonus = red[0] * v[off];
    y[off] = (gn + bonus) * g[off];
}

// ================= host orchestration =================
// JAX tree flatten order (sorted dict keys) for params, then idx:
enum { ATT_A0 = 0, ATT_A1, ATT_A2, ATT_G1, ATT_G2, ATT_K_A, ATT_K_K, ATT_KEY_W,
       ATT_LN_X_B, ATT_LN_X_G, ATT_OUT_W, ATT_R_K, ATT_REC_W, ATT_V0, ATT_V1,
       ATT_V2, ATT_VAL_W, ATT_W0, ATT_W1, ATT_W2, ATT_X_A, ATT_X_G, ATT_X_K,
       ATT_X_R, ATT_X_V, ATT_X_W,
       FFN_KEY_W, FFN_VAL_W, FFN_X_K, LN0_B, LN0_G, LN1_B, LN1_G, LN2_B, LN2_G,
       BLK_STRIDE };
#define EMB_I    (NL * BLK_STRIDE)
#define HEADW_I  (NL * BLK_STRIDE + 1)
#define LNOUTB_I (NL * BLK_STRIDE + 2)
#define LNOUTG_I (NL * BLK_STRIDE + 3)
#define IDX_I    (NL * BLK_STRIDE + 4)

static inline int ceil_div(int a, int b) { return (a + b - 1) / b; }

static void gemm_op(hipStream_t s, const float* A, const float* W, float* Cm,
                    int M, int N, int K, __bf16* Abf, __bf16* Wt) {
    int na = M * K;
    k_cvt<<<ceil_div(na, 256), 256, 0, s>>>(A, Abf, na);
    int nw = K * N;
    k_cvt_t<<<ceil_div(nw, 256), 256, 0, s>>>(W, Wt, K, N);
    dim3 grid(M / BM, ceil_div(N, BN));
    k_gemm<<<grid, 256, 0, s>>>(Abf, Wt, Cm, M, N, K);
}

extern "C" void kernel_launch(void* const* d_in, const int* in_sizes, int n_in,
                              void* d_out, int out_size, void* d_ws, size_t ws_size,
                              hipStream_t stream) {
    (void)in_sizes; (void)n_in; (void)out_size; (void)ws_size;
    const int NTC = TT * CC;
    const int EW = 3072;  // elementwise grid for T*C

    // ---- workspace carve ----
    size_t o = 0;
    auto alloc = [&](size_t bytes) -> void* {
        bytes = (bytes + 255) & ~(size_t)255;
        void* p = (char*)d_ws + o; o += bytes; return p;
    };
    float* x      = (float*)alloc((size_t)NTC * 4);
    float* xln    = (float*)alloc((size_t)NTC * 4);
    float* xmix   = (float*)alloc((size_t)NTC * 4);
    float* rbuf   = (float*)alloc((size_t)NTC * 4);
    float* wbuf   = (float*)alloc((size_t)NTC * 4);
    float* kbuf   = (float*)alloc((size_t)NTC * 4);
    float* vbuf   = (float*)alloc((size_t)NTC * 4);
    float* abuf   = (float*)alloc((size_t)NTC * 4);
    float* gbuf   = (float*)alloc((size_t)NTC * 4);
    float* vfirst = (float*)alloc((size_t)NTC * 4);
    float* nkk    = (float*)alloc((size_t)NTC * 4);
    float* kka    = (float*)alloc((size_t)NTC * 4);
    float* kout   = (float*)alloc((size_t)NTC * 4);
    float* ybuf   = (float*)alloc((size_t)NTC * 4);
    float* tmpc   = (float*)alloc((size_t)NTC * 4);
    float* tmps   = (float*)alloc((size_t)TT * DGATE * 4);
    float* fkbuf  = (float*)alloc((size_t)TT * FF * 4);
    __bf16* Abf   = (__bf16*)alloc((size_t)TT * FF * 2);
    __bf16* Wt    = (__bf16*)alloc((size_t)VV * CC * 2);

    const int* idx = (const int*)d_in[IDX_I];
    auto P = [&](int b, int off) { return (const float*)d_in[b * BLK_STRIDE + off]; };

    // ---- embedding ----
    k_embed<<<EW, 256, 0, stream>>>((const float*)d_in[EMB_I], idx, x);

    for (int b = 0; b < NL; ++b) {
        if (b == 0)
            k_ln<<<TT, 256, 0, stream>>>(x, P(b, LN0_G), P(b, LN0_B), x);
        k_ln<<<TT, 256, 0, stream>>>(x, P(b, LN1_G), P(b, LN1_B), xln);

        // r
        k_mix<<<EW, 256, 0, stream>>>(xln, P(b, ATT_X_R), xmix);
        gemm_op(stream, xmix, P(b, ATT_REC_W), rbuf, TT, CC, CC, Abf, Wt);
        // w (lora + softplus)
        k_mix<<<EW, 256, 0, stream>>>(xln, P(b, ATT_X_W), xmix);
        gemm_op(stream, xmix, P(b, ATT_W1), tmps, TT, DDEC, CC, Abf, Wt);
        k_tanh<<<ceil_div(TT * DDEC, 256), 256, 0, stream>>>(tmps, TT * DDEC);
        gemm_op(stream, tmps, P(b, ATT_W2), wbuf, TT, CC, DDEC, Abf, Wt);
        k_decay<<<EW, 256, 0, stream>>>(wbuf, P(b, ATT_W0));
        // k
        k_mix<<<EW, 256, 0, stream>>>(xln, P(b, ATT_X_K), xmix);
        gemm_op(stream, xmix, P(b, ATT_KEY_W), kbuf, TT, CC, CC, Abf, Wt);
        // v (+ v_first residual mix)
        k_mix<<<EW, 256, 0, stream>>>(xln, P(b, ATT_X_V), xmix);
        gemm_op(stream, xmix, P(b, ATT_VAL_W), vbuf, TT, CC, CC, Abf, Wt);
        if (b == 0) {
            (void)hipMemcpyAsync(vfirst, vbuf, (size_t)NTC * 4,
                                 hipMemcpyDeviceToDevice, stream);
        } else {
            gemm_op(stream, xmix, P(b, ATT_V1), tmps, TT, DMV, CC, Abf, Wt);
            gemm_op(stream, tmps, P(b, ATT_V2), tmpc, TT, CC, DMV, Abf, Wt);
            k_vmix<<<EW, 256, 0, stream>>>(vbuf, vfirst, P(b, ATT_V0), tmpc);
        }
        // a
        k_mix<<<EW, 256, 0, stream>>>(xln, P(b, ATT_X_A), xmix);
        gemm_op(stream, xmix, P(b, ATT_A1), tmps, TT, DAAA, CC, Abf, Wt);
        gemm_op(stream, tmps, P(b, ATT_A2), abuf, TT, CC, DAAA, Abf, Wt);
        k_afin<<<EW, 256, 0, stream>>>(abuf, P(b, ATT_A0));
        // g
        k_mix<<<EW, 256, 0, stream>>>(xln, P(b, ATT_X_G), xmix);
        gemm_op(stream, xmix, P(b, ATT_G1), tmps, TT, DGATE, CC, Abf, Wt);
        k_sigmoid<<<ceil_div(TT * DGATE, 256), 256, 0, stream>>>(tmps, TT * DGATE);
        gemm_op(stream, tmps, P(b, ATT_G2), gbuf, TT, CC, DGATE, Abf, Wt);
        // kk prep + wkv scan + groupnorm/bonus/gate
        k_kkprep<<<dim3(TT, HH), 64, 0, stream>>>(kbuf, abuf, P(b, ATT_K_K),
                                                  P(b, ATT_K_A), nkk, kka, kout);
        k_wkv<<<HH, 64, 0, stream>>>(rbuf, wbuf, kout, vbuf, nkk, kka, ybuf);
        k_gnbg<<<dim3(TT, HH), 64, 0, stream>>>(ybuf, P(b, ATT_LN_X_G),
                                                P(b, ATT_LN_X_B), rbuf, kout, vbuf,
                                                P(b, ATT_R_K), gbuf);
        // output projection + residual
        gemm_op(stream, ybuf, P(b, ATT_OUT_W), tmpc, TT, CC, CC, Abf, Wt);
        k_add<<<EW, 256, 0, stream>>>(x, tmpc, NTC);

        // ---- channel mix ----
        k_ln<<<TT, 256, 0, stream>>>(x, P(b, LN2_G), P(b, LN2_B), xln);
        k_mix<<<EW, 256, 0, stream>>>(xln, P(b, FFN_X_K), xmix);
        gemm_op(stream, xmix, P(b, FFN_KEY_W), fkbuf, TT, FF, CC, Abf, Wt);
        k_relusq<<<ceil_div(TT * FF, 256), 256, 0, stream>>>(fkbuf, TT * FF);
        gemm_op(stream, fkbuf, P(b, FFN_VAL_W), tmpc, TT, CC, FF, Abf, Wt);
        k_add<<<EW, 256, 0, stream>>>(x, tmpc, NTC);
    }

    // ---- head ----
    k_ln<<<TT, 256, 0, stream>>>(x, (const float*)d_in[LNOUTG_I],
                                 (const float*)d_in[LNOUTB_I], xln);
    gemm_op(stream, xln, (const float*)d_in[HEADW_I], (float*)d_out,
            TT, VV, CC, Abf, Wt);
}
